// MT_46660524703791
// MI455X (gfx1250) — matrix operand order repaired and checked by reading, not compiled, not run on hardware
//
#include <hip/hip_runtime.h>
#include <hip/hip_bf16.h>
#include <stdint.h>

// Problem dims (compile-time constants from the reference)
#define BB      16384
#define IN_DIM  1024
#define H1D     2048
#define H2D     2048
#define HIDD    1024
#define SPECH   2048
#define NEXP    8

// LDS tile strides, padded for conflict-free (2-way, optimal for b128) banking
#define ATS   80     // A tile row stride bytes (64 data + 16 pad)
#define BTS   272    // B tile row stride bytes (256 data + 16 pad)
#define ABYT  (128 * ATS)          // 10240
#define BBYT  (32 * BTS)           // 8704
#define BUFB  (ABYT + BBYT)        // 18944 per pipeline buffer

// WMMA operand types (ext_vector, verified lowering)
typedef __attribute__((ext_vector_type(16))) __bf16 v16bf;
typedef __attribute__((ext_vector_type(8)))  float  v8f;

// Builtin argument types: GCC-vector form (matches builtin declarations)
typedef int          v4i   __attribute__((__vector_size__(16)));
typedef unsigned int u32x4 __attribute__((__vector_size__(16)));
typedef int          i32x8 __attribute__((__vector_size__(32)));
typedef int          i32x4 __attribute__((__vector_size__(16)));

#define AS1 __attribute__((address_space(1)))
#define AS3 __attribute__((address_space(3)))

#if defined(__has_builtin)
#if __has_builtin(__builtin_amdgcn_tensor_load_to_lds)
#define HAVE_TDM 1
#endif
#if __has_builtin(__builtin_amdgcn_global_load_async_to_lds_b128)
#define HAVE_ASYNC 1
#endif
#endif
#ifndef HAVE_TDM
#define HAVE_TDM 0
#endif
#ifndef HAVE_ASYNC
#define HAVE_ASYNC 0
#endif

union Frag { uint4 u[2]; v16bf v; };

// Low 32 bits of a flat shared-memory pointer are the LDS byte offset.
__device__ __forceinline__ uint32_t lds_byte_off(const void* p) {
  return (uint32_t)(uintptr_t)p;
}

#if HAVE_ASYNC
__device__ __forceinline__ void async_b128(const void* g, void* l) {
  __builtin_amdgcn_global_load_async_to_lds_b128(
      (AS1 v4i*)(uintptr_t)g,
      (AS3 v4i*)(uintptr_t)lds_byte_off(l), 0, 0);
}
__device__ __forceinline__ void wait_async0() {
#if __has_builtin(__builtin_amdgcn_s_wait_asynccnt)
  __builtin_amdgcn_s_wait_asynccnt(0);
#else
  asm volatile("s_wait_asynccnt 0x0" ::: "memory");
#endif
}
#endif

#if HAVE_TDM
// One-shot TDM 2D tile load: tileRows x tileCols bf16 from row-major global
// (rowStrideElems) into LDS at ldsByte, with optional LDS row padding
// (padInterval/padAmount encodings per D# group1; 0/0 = disabled).
__device__ __forceinline__ void tdm_load_2d_bf16(const __bf16* gsrc, uint32_t ldsByte,
                                                 uint32_t tileCols, uint32_t tileRows,
                                                 uint32_t rowStrideElems,
                                                 uint32_t padInterval, uint32_t padAmount) {
  uint64_t ga = (uint64_t)(uintptr_t)gsrc;
  u32x4 g0;
  g0[0] = 1u;                                                 // count=1, user mode
  g0[1] = ldsByte;                                            // lds_addr
  g0[2] = (uint32_t)ga;                                       // global_addr[31:0]
  g0[3] = (uint32_t)((ga >> 32) & 0x01FFFFFFu) | (2u << 30);  // addr[56:32] | type=2
  i32x8 g1;
  uint32_t d0 = rowStrideElems;       // tensor_dim0 (elems) -> tile always in-bounds
  uint32_t d1 = 0x40000000u;          // tensor_dim1: effectively unbounded
  uint32_t w0 = (1u << 16);           // data_size=1 (2 bytes/elem)
  if (padAmount) w0 |= (1u << 20) | (padInterval << 22) | (padAmount << 25);
  g1[0] = (int)w0;
  g1[1] = (int)((d0 & 0xFFFFu) << 16);                 // tensor_dim0[15:0]
  g1[2] = (int)((d0 >> 16) | ((d1 & 0xFFFFu) << 16));  // dim0[31:16] | dim1[15:0]
  g1[3] = (int)((d1 >> 16) | (tileCols << 16));        // dim1[31:16] | tile_dim0
  g1[4] = (int)(tileRows & 0xFFFFu);                   // tile_dim1 (tile_dim2=0)
  g1[5] = (int)rowStrideElems;                         // tensor_dim0_stride[31:0]
  g1[6] = 0;
  g1[7] = 0;
  i32x4 z4 = {0, 0, 0, 0};
  i32x8 z8 = {0, 0, 0, 0, 0, 0, 0, 0};
  __builtin_amdgcn_tensor_load_to_lds(g0, g1, z4, z4, z8, 0);
}
#endif

// ---------------------------------------------------------------------------
// fp32 -> bf16 elementwise conversion (8 elems/thread, b128 in/out)
// ---------------------------------------------------------------------------
__global__ __launch_bounds__(256) void mt_cvt_bf16(const float* __restrict__ src,
                                                   __bf16* __restrict__ dst, int n) {
  int i = (blockIdx.x * 256 + threadIdx.x) * 8;
  if (i >= n) return;
  float4 f0 = *(const float4*)(src + i);
  float4 f1 = *(const float4*)(src + i + 4);
  __bf16 t[8];
  t[0] = (__bf16)f0.x; t[1] = (__bf16)f0.y; t[2] = (__bf16)f0.z; t[3] = (__bf16)f0.w;
  t[4] = (__bf16)f1.x; t[5] = (__bf16)f1.y; t[6] = (__bf16)f1.z; t[7] = (__bf16)f1.w;
  *(uint4*)(dst + i) = *(uint4*)t;
}

// ---------------------------------------------------------------------------
// Router: bucket rows by expert id (bucket order is atomic-dependent but each
// row's result is order-independent, so output is deterministic).
// ---------------------------------------------------------------------------
__global__ __launch_bounds__(256) void mt_route(const int* __restrict__ d,
                                                int* __restrict__ counts,
                                                int* __restrict__ rowlist) {
  int i = blockIdx.x * 256 + threadIdx.x;
  if (i < BB) {
    int e = d[i];
    if (e < 0 || e >= NEXP) e = 0;
    int slot = atomicAdd(&counts[e], 1);
    rowlist[e * BB + slot] = i;
  }
}

// ---------------------------------------------------------------------------
// Double-buffered WMMA GEMM: C[M,N] = act(A[M,K] @ W[K,N] + b), all bf16 in.
//   256 threads / 8 waves; 128x128 C tile; wave = 64x32 (4x2 WMMA accums).
//   LDS tiles are row-major with padded strides (A 128x32 @80B, B 32x128
//   @272B) -> fragment ds_load_b128s are 2-way bank-balanced (optimal).
//   Staging: TDM tensor_load_to_lds (pad inserted by the DMA engine), or
//   per-lane async-to-LDS / load+store fallbacks. GATHER stages A through
//   rowlist.
// ---------------------------------------------------------------------------
template <bool GATHER, bool RELU>
__global__ __launch_bounds__(256) void mt_gemm(
    const __bf16* __restrict__ A, const __bf16* __restrict__ Wall,
    const float* __restrict__ ball, __bf16* __restrict__ Cout,
    int K, int N,
    const int* __restrict__ rowlistAll, const int* __restrict__ counts) {
  __shared__ __align__(16) char smem[2 * BUFB];  // pipeline bufs; epilogue reuses 32KB

  const int t = threadIdx.x;
  const int lane = t & 31;
  const int wv = t >> 5;
  const int bx = blockIdx.x, by = blockIdx.y, ex = blockIdx.z;

  const __bf16* Wm = Wall;
  const float*  bm = ball;
  const int*    rl = nullptr;
  int Me = BB;
  if (GATHER) {
    Wm = Wall + (size_t)ex * (size_t)K * (size_t)N;
    bm = ball + (size_t)ex * (size_t)N;
    rl = rowlistAll + (size_t)ex * (size_t)BB;
    Me = counts[ex];
    if (by * 128 >= Me) return;
  }

  // Per-thread staging coordinates (per-lane paths)
  const int sr = t >> 1, sh = t & 1;     // A: row 0..127, 32B half
  const int bk = t >> 3, bseg = t & 7;   // B: row 0..31, 32B segment

  int  agrow  = by * 128 + sr;
  bool avalid = true;
  if (GATHER) { avalid = agrow < Me; agrow = avalid ? rl[agrow] : 0; }
  const __bf16* aRow     = A + (size_t)agrow * K + sh * 16;
  const __bf16* bRowBase = Wm + (size_t)bk * N + bx * 128 + bseg * 16;

  v8f acc[4][2] = {};
  const int wm  = wv >> 2, wn = wv & 3;
  const int am  = lane & 15;
  const int hk2 = (lane >> 4) * 16;                  // A frag K-half byte offset
  const int bkk = (lane & 15) + ((lane >> 4) << 4);  // B frag K row

  auto stage = [&](int k0, int buf) {
    char* ab = smem + buf * BUFB;
    char* bb = ab + ABYT;
#if HAVE_TDM
    if (wv == 0) {
      if (!GATHER)  // A: 128 rows x 32 elems, pad 16B per 64B row
        tdm_load_2d_bf16(A + (size_t)(by * 128) * K + k0, lds_byte_off(ab),
                         32, 128, K, /*interval 64B*/ 3, /*pad 16B*/ 3);
      // B: 32 rows x 128 elems, pad 16B per 256B row
      tdm_load_2d_bf16(Wm + (size_t)k0 * N + bx * 128, lds_byte_off(bb),
                       128, 32, N, /*interval 256B*/ 5, /*pad 16B*/ 3);
    }
    if (GATHER) {
      char* ad = ab + sr * ATS + sh * 32;
      const __bf16* as = aRow + k0;
      if (avalid) {
#if HAVE_ASYNC
        async_b128(as, ad); async_b128(as + 8, ad + 16);
#else
        ((uint4*)ad)[0] = ((const uint4*)as)[0];
        ((uint4*)ad)[1] = ((const uint4*)as)[1];
#endif
      } else {
        uint4 zq = {0, 0, 0, 0};
        ((uint4*)ad)[0] = zq; ((uint4*)ad)[1] = zq;
      }
    }
#else  // no TDM: stage both tiles per-lane (async if available)
    {
      char* ad = ab + sr * ATS + sh * 32;
      const __bf16* as = aRow + k0;
      if (!GATHER || avalid) {
#if HAVE_ASYNC
        async_b128(as, ad); async_b128(as + 8, ad + 16);
#else
        ((uint4*)ad)[0] = ((const uint4*)as)[0];
        ((uint4*)ad)[1] = ((const uint4*)as)[1];
#endif
      } else {
        uint4 zq = {0, 0, 0, 0};
        ((uint4*)ad)[0] = zq; ((uint4*)ad)[1] = zq;
      }
    }
    {
      char* bd = bb + bk * BTS + bseg * 32;
      const __bf16* bs = bRowBase + (size_t)k0 * N;
#if HAVE_ASYNC
      async_b128(bs, bd); async_b128(bs + 8, bd + 16);
#else
      ((uint4*)bd)[0] = ((const uint4*)bs)[0];
      ((uint4*)bd)[1] = ((const uint4*)bs)[1];
#endif
    }
#endif
  };

  auto stage_wait = [&]() {
#if HAVE_TDM
    if (wv == 0) __builtin_amdgcn_s_wait_tensorcnt(0);
#if HAVE_ASYNC
    if (GATHER) wait_async0();
#endif
#else
#if HAVE_ASYNC
    wait_async0();
#endif
#endif
    __syncthreads();
  };

  auto compute = [&](int buf) {
    const char* ab = smem + buf * BUFB;
    const char* bb = ab + ABYT;
    Frag fa[4], fb[2];
#pragma unroll
    for (int i = 0; i < 4; ++i) {
      const char* p = ab + ((wm * 4 + i) * 16 + am) * ATS + hk2;
      fa[i].u[0] = *(const uint4*)p;
      fa[i].u[1] = *(const uint4*)(p + 32);
    }
#pragma unroll
    for (int j = 0; j < 2; ++j) {
      const char* p = bb + ((wn * 2 + j) * 16 + (lane & 15)) * 2;
#pragma unroll
      for (int s = 0; s < 16; ++s) {
        const int kk = (s < 8) ? ((lane >> 4) * 8 + s) : (16 + (lane >> 4) * 8 + (s - 8));
        fb[j].v[s] = *(const __bf16*)(p + kk * BTS);
      }
    }
#pragma unroll
    for (int i = 0; i < 4; ++i)
#pragma unroll
      for (int j = 0; j < 2; ++j)
        acc[i][j] = __builtin_amdgcn_wmma_f32_16x16x32_bf16(
            false, fa[i].v, false, fb[j].v, (short)0, acc[i][j], false, false);
  };

  // Software-pipelined main loop: stage slab k+1 while computing slab k.
  stage(0, 0);
  stage_wait();
  int buf = 0;
  for (int k0 = 0; k0 < K; k0 += 32) {
    if (k0 + 32 < K) stage(k0 + 32, buf ^ 1);
    compute(buf);
    stage_wait();
    buf ^= 1;
  }

  // Epilogue: bias(+ReLU) -> bf16 through LDS -> coalesced b128 stores
  __bf16* cS = (__bf16*)smem;
#pragma unroll
  for (int j = 0; j < 2; ++j) {
    const int   nl = (wn * 2 + j) * 16 + (lane & 15);
    const float bv = bm[bx * 128 + nl];
#pragma unroll
    for (int i = 0; i < 4; ++i) {
      const int ml = (wm * 4 + i) * 16 + (lane >> 4) * 8;
#pragma unroll
      for (int g = 0; g < 8; ++g) {
        float v = acc[i][j][g] + bv;
        if (RELU) v = fmaxf(v, 0.f);
        cS[(ml + g) * 128 + nl] = (__bf16)v;
      }
    }
  }
  __syncthreads();
  if (!GATHER || avalid) {
    const uint4* s  = (const uint4*)(cS + sr * 128 + sh * 64);
    uint4*       dp = (uint4*)(Cout + (size_t)agrow * N + bx * 128 + sh * 64);
#pragma unroll
    for (int q = 0; q < 8; ++q) dp[q] = s[q];
  }
}

// ---------------------------------------------------------------------------
// Head layer 2 (OUT=1): wave-per-row GEMV  out[i] = h[i,:] . W2[d[i]] + b2[d[i]]
// ---------------------------------------------------------------------------
__global__ __launch_bounds__(256) void mt_head2(const __bf16* __restrict__ h,
                                                const float* __restrict__ W2,
                                                const float* __restrict__ b2,
                                                const int* __restrict__ d,
                                                float* __restrict__ out) {
  const int lane = threadIdx.x & 31;
  const int wv   = threadIdx.x >> 5;
  const int row  = blockIdx.x * 8 + wv;
  if (row >= BB) return;
  int e = d[row];
  if (e < 0 || e >= NEXP) e = 0;
  const __bf16* hp = h + (size_t)row * SPECH;
  const float*  wp = W2 + (size_t)e * SPECH;
  float acc = 0.f;
#pragma unroll 4
  for (int c = 0; c < SPECH / 64; ++c) {
    const int k = c * 64 + lane * 2;
    __bf16 h0 = hp[k];
    __bf16 h1 = hp[k + 1];
    float2 w  = *(const float2*)(wp + k);
    acc += (float)h0 * w.x + (float)h1 * w.y;
  }
#pragma unroll
  for (int off = 16; off > 0; off >>= 1) acc += __shfl_xor(acc, off, 32);
  if (lane == 0) out[row] = acc + b2[e];
}

// ---------------------------------------------------------------------------
extern "C" void kernel_launch(void* const* d_in, const int* in_sizes, int n_in,
                              void* d_out, int out_size, void* d_ws, size_t ws_size,
                              hipStream_t stream) {
  (void)in_sizes; (void)n_in; (void)out_size; (void)ws_size;
  const float* x   = (const float*)d_in[0];
  const int*   d   = (const int*)d_in[1];
  const float* W0  = (const float*)d_in[2];
  const float* b0  = (const float*)d_in[3];
  const float* W1  = (const float*)d_in[4];
  const float* b1  = (const float*)d_in[5];
  const float* W2  = (const float*)d_in[6];
  const float* b2  = (const float*)d_in[7];
  const float* sW1 = (const float*)d_in[8];
  const float* sb1 = (const float*)d_in[9];
  const float* sW2 = (const float*)d_in[10];
  const float* sb2 = (const float*)d_in[11];
  float* out = (float*)d_out;

  const size_t MB = (size_t)1 << 20;
  char*   ws      = (char*)d_ws;
  __bf16* a1      = (__bf16*)(ws);               // [B,H1]  64MB (reused as h)
  __bf16* a2      = (__bf16*)(ws + 64 * MB);     // [B,H2]  64MB
  __bf16* xb      = a2;                          // xb (32MB) overlays a2 (dead before L2)
  __bf16* z       = (__bf16*)(ws + 128 * MB);    // [B,HID] 32MB
  __bf16* w0b     = (__bf16*)(ws + 160 * MB);    // 4MB
  __bf16* w1b     = (__bf16*)(ws + 164 * MB);    // 8MB
  __bf16* w2b     = (__bf16*)(ws + 172 * MB);    // 4MB
  __bf16* sw1b    = (__bf16*)(ws + 176 * MB);    // 32MB
  int*    rowlist = (int*)(ws + 208 * MB);       // [E,B] 512KB
  int*    counts  = rowlist + (size_t)NEXP * BB; // [E]
  __bf16* h       = a1;

  // Router
  (void)hipMemsetAsync(counts, 0, NEXP * sizeof(int), stream);
  mt_route<<<BB / 256, 256, 0, stream>>>(d, counts, rowlist);

  // One-time fp32 -> bf16 conversion of activations-in and weights
  auto cvt = [&](const float* s, __bf16* dst, int n) {
    mt_cvt_bf16<<<n / 2048, 256, 0, stream>>>(s, dst, n);
  };
  cvt(x,   xb,   BB * IN_DIM);
  cvt(W0,  w0b,  IN_DIM * H1D);
  cvt(W1,  w1b,  H1D * H2D);
  cvt(W2,  w2b,  H2D * HIDD);
  cvt(sW1, sw1b, NEXP * HIDD * SPECH);

  dim3 blk(256);
  // Encoder
  mt_gemm<false, true><<<dim3(H1D / 128, BB / 128, 1), blk, 0, stream>>>(
      xb, w0b, b0, a1, IN_DIM, H1D, nullptr, nullptr);
  mt_gemm<false, true><<<dim3(H2D / 128, BB / 128, 1), blk, 0, stream>>>(
      a1, w1b, b1, a2, H1D, H2D, nullptr, nullptr);
  mt_gemm<false, false><<<dim3(HIDD / 128, BB / 128, 1), blk, 0, stream>>>(
      a2, w2b, b2, z, H2D, HIDD, nullptr, nullptr);
  // Routed expert layer 1 (row-gathered per expert)
  mt_gemm<true, true><<<dim3(SPECH / 128, BB / 128, NEXP), blk, 0, stream>>>(
      z, sw1b, sb1, h, HIDD, SPECH, rowlist, counts);
  // Routed expert layer 2 (OUT=1 GEMV)
  mt_head2<<<BB / 8, 256, 0, stream>>>(h, sW2, sb2, d, out);
}
